// Transformer_61632780697909
// MI455X (gfx1250) — compile-verified, matrix operand order checked by reading, not run on hardware
//
#include <hip/hip_runtime.h>
#include <hip/hip_bf16.h>

typedef __attribute__((ext_vector_type(16))) _Float16 v16h;
typedef __attribute__((ext_vector_type(8)))  _Float16 v8h;
typedef __attribute__((ext_vector_type(8)))  float    v8f;

union HV { v16h v; v8h h[2]; };

#define TB  4
#define TS  2048
#define TD  768
#define TH  12
#define TDK 64
#define TF  3072
#define TL  6
#define TM  (TB*TS)      // 8192 rows
#define QKVN (3*TD)      // 2304 fused QKV width

// ---------------------------------------------------------------------------
// f32 [L][K][N] -> f16 transposed [L][rowOff+n][K] (weights staged as B^T so
// the GEMM hot loop does contiguous b128 copies, no scalar LDS transpose)
// ---------------------------------------------------------------------------
__global__ __launch_bounds__(256)
void cvt_transpose_kernel(const float* __restrict__ src, _Float16* __restrict__ dst,
                          int K, int N, int layerStride, int rowOff, int total) {
  int idx = blockIdx.x * 256 + threadIdx.x;
  if (idx >= total) return;               // total = L*N*K
  int nk = N * K;
  int l = idx / nk;
  int rem = idx - l * nk;
  int n = rem / K;
  int k = rem - n * K;
  dst[(size_t)l * layerStride + (size_t)(rowOff + n) * K + k] =
      (_Float16)src[((size_t)l * K + k) * N + n];
}

// ---------------------------------------------------------------------------
// Concatenated QKV bias: [L][2304]
// ---------------------------------------------------------------------------
__global__ __launch_bounds__(256)
void catbias_kernel(const float* __restrict__ bq, const float* __restrict__ bk,
                    const float* __restrict__ bv, float* __restrict__ dst) {
  int idx = blockIdx.x * 256 + threadIdx.x;
  if (idx >= TL * QKVN) return;
  int l = idx / QKVN, j = idx - l * QKVN;
  float v = (j < TD) ? bq[l * TD + j]
          : (j < 2 * TD) ? bk[l * TD + j - TD]
          : bv[l * TD + j - 2 * TD];
  dst[idx] = v;
}

// ---------------------------------------------------------------------------
// Embedding + positional encoding + key mask
// ---------------------------------------------------------------------------
__global__ __launch_bounds__(256) void embed_kernel(const int* __restrict__ ids,
                                                    const float* __restrict__ emb,
                                                    float* __restrict__ xf,
                                                    _Float16* __restrict__ xh,
                                                    float* __restrict__ maskf) {
  int idx = blockIdx.x * 256 + threadIdx.x;
  if (idx >= TM * TD) return;
  int d = idx % TD;
  int t = idx / TD;
  int s = t % TS;
  int id = ids[t];
  float i2   = (float)(d & ~1);
  float freq = __expf(-9.210340371976184f * i2 / (float)TD);  // ln(10000)
  float ang  = (float)s * freq;
  float pe   = (d & 1) ? cosf(ang) : sinf(ang);
  float v = emb[(size_t)id * TD + d] + pe;
  xf[idx] = v;
  xh[idx] = (_Float16)v;
  if (d == 0) maskf[t] = (id != 1) ? 0.0f : -1e9f;
}

// ---------------------------------------------------------------------------
// WMMA GEMM v2: C = A[MxK] * Bt^T + bias (Bt is [N][K] f16), optional ReLU.
// 128x128 block, 256 thr = 8 waves (4x2), wave tile 32x64 -> 8 WMMA / K-step.
// Double-buffered LDS: global->reg prefetch overlaps WMMA, reg->LDS after.
// ---------------------------------------------------------------------------
__global__ __launch_bounds__(256)
void gemm_f16_kernel(const _Float16* __restrict__ A, const _Float16* __restrict__ Bt,
                     const float* __restrict__ bias, float* __restrict__ Cf32,
                     _Float16* __restrict__ Cf16, int M, int N, int K, int relu) {
  __shared__ __align__(16) _Float16 sA[2][128 * 32];   // [row][k]
  __shared__ __align__(16) _Float16 sB[2][128 * 32];   // [n][k]
  int tid  = threadIdx.x;
  int wave = tid >> 5, lane = tid & 31;
  int wm = wave >> 1, wn = wave & 1;
  int laneN = lane & 15, laneHi = lane >> 4;
  int kbo = laneHi * 8;
  int m0 = blockIdx.y * 128, n0 = blockIdx.x * 128;

  // staging coords: each thread copies rows r0 and r0+64, 8 halves at kc
  int r0 = tid >> 2, kc = (tid & 3) * 8;
  const _Float16* gA = A  + (size_t)(m0 + r0) * K + kc;
  const _Float16* gB = Bt + (size_t)(n0 + r0) * K + kc;
  const size_t rstep = (size_t)64 * K;

  // prologue: stage K-step 0 into buffer 0
  {
    v8h a0 = *(const v8h*)gA;
    v8h a1 = *(const v8h*)(gA + rstep);
    v8h b0 = *(const v8h*)gB;
    v8h b1 = *(const v8h*)(gB + rstep);
    *(v8h*)(&sA[0][r0 * 32 + kc])        = a0;
    *(v8h*)(&sA[0][(r0 + 64) * 32 + kc]) = a1;
    *(v8h*)(&sB[0][r0 * 32 + kc])        = b0;
    *(v8h*)(&sB[0][(r0 + 64) * 32 + kc]) = b1;
  }

  v8f zero = {};
  v8f c[2][4];
#pragma unroll
  for (int i = 0; i < 2; ++i)
#pragma unroll
    for (int j = 0; j < 4; ++j) c[i][j] = zero;

  int cur = 0;
  for (int k0 = 0; k0 < K; k0 += 32) {
    __syncthreads();
    bool next = (k0 + 32) < K;
    v8h na0 = {}, na1 = {}, nb0 = {}, nb1 = {};
    if (next) {                              // uniform branch (no EXEC change)
      const _Float16* pA = gA + k0 + 32;
      const _Float16* pB = gB + k0 + 32;
      na0 = *(const v8h*)pA;
      na1 = *(const v8h*)(pA + rstep);
      nb0 = *(const v8h*)pB;
      nb1 = *(const v8h*)(pB + rstep);
      __builtin_prefetch(pA + 32, 0, 1);     // global_prefetch_b8
      __builtin_prefetch(pB + 32, 0, 1);
    }

    const _Float16* bufA = sA[cur];
    const _Float16* bufB = sB[cur];
    HV af[2], bf[4];
#pragma unroll
    for (int i = 0; i < 2; ++i) {
      int row = wm * 32 + i * 16 + laneN;
      af[i].h[0] = *(const v8h*)(bufA + row * 32 + kbo);
      af[i].h[1] = *(const v8h*)(bufA + row * 32 + 16 + kbo);
    }
#pragma unroll
    for (int j = 0; j < 4; ++j) {
      int col = wn * 64 + j * 16 + laneN;
      bf[j].h[0] = *(const v8h*)(bufB + col * 32 + kbo);
      bf[j].h[1] = *(const v8h*)(bufB + col * 32 + 16 + kbo);
    }
#pragma unroll
    for (int i = 0; i < 2; ++i)
#pragma unroll
      for (int j = 0; j < 4; ++j)
        c[i][j] = __builtin_amdgcn_wmma_f32_16x16x32_f16(false, af[i].v, false, bf[j].v,
                                                         (short)0, c[i][j], false, false);
    if (next) {
      int nx = cur ^ 1;
      *(v8h*)(&sA[nx][r0 * 32 + kc])        = na0;
      *(v8h*)(&sA[nx][(r0 + 64) * 32 + kc]) = na1;
      *(v8h*)(&sB[nx][r0 * 32 + kc])        = nb0;
      *(v8h*)(&sB[nx][(r0 + 64) * 32 + kc]) = nb1;
    }
    cur ^= 1;
  }

  // epilogue: C layout -> lane N = laneN, M = base + laneHi*8 + r
#pragma unroll
  for (int i = 0; i < 2; ++i) {
    int rowBase = m0 + wm * 32 + i * 16 + laneHi * 8;
#pragma unroll
    for (int j = 0; j < 4; ++j) {
      int col = n0 + wn * 64 + j * 16 + laneN;
      float bb = bias ? bias[col] : 0.0f;
#pragma unroll
      for (int r = 0; r < 8; ++r) {
        float v = c[i][j][r] + bb;
        if (relu) v = fmaxf(v, 0.0f);
        size_t idx = (size_t)(rowBase + r) * N + col;
        if (Cf32) Cf32[idx] = v;
        if (Cf16) Cf16[idx] = (_Float16)v;
      }
    }
  }
}

// ---------------------------------------------------------------------------
// Flash attention over fused QKV buffer [M][2304] (q | k | v per head).
// grid (S/64, H, B), 128 thr = 4 waves; each wave = 16-row query tile.
// ---------------------------------------------------------------------------
__device__ __forceinline__ float red_max16(float v) {
  v = fmaxf(v, __shfl_xor(v, 1, 32));
  v = fmaxf(v, __shfl_xor(v, 2, 32));
  v = fmaxf(v, __shfl_xor(v, 4, 32));
  v = fmaxf(v, __shfl_xor(v, 8, 32));
  return v;
}
__device__ __forceinline__ float red_sum16(float v) {
  v += __shfl_xor(v, 1, 32);
  v += __shfl_xor(v, 2, 32);
  v += __shfl_xor(v, 4, 32);
  v += __shfl_xor(v, 8, 32);
  return v;
}

__global__ __launch_bounds__(128)
void attn_kernel(const _Float16* __restrict__ QKV, const float* __restrict__ maskf,
                 _Float16* __restrict__ O) {
  __shared__ __align__(16) _Float16 sK[32 * 64];        // [key][dk]
  __shared__ __align__(16) _Float16 sV[64 * 32];        // [dk][key] (transposed)
  __shared__ __align__(16) _Float16 sP[4][16 * 32];     // per-wave P [row][key]

  int tid = threadIdx.x;
  int wave = tid >> 5, lane = tid & 31;
  int laneN = lane & 15, laneHi = lane >> 4;
  int kb = laneHi * 8;
  int h = blockIdx.y, b = blockIdx.z;
  int qBase = blockIdx.x * 64 + wave * 16;

  const _Float16* qrow = QKV + ((size_t)b * TS + qBase + laneN) * QKVN + h * TDK;
  HV q0, q1;
  q0.h[0] = *(const v8h*)(qrow + kb);
  q0.h[1] = *(const v8h*)(qrow + 16 + kb);
  q1.h[0] = *(const v8h*)(qrow + 32 + kb);
  q1.h[1] = *(const v8h*)(qrow + 32 + 16 + kb);

  v8f o0 = {}, o1 = {}, o2 = {}, o3 = {};
  float mrow[8], lrow[8];
#pragma unroll
  for (int r = 0; r < 8; ++r) { mrow[r] = -1e30f; lrow[r] = 0.0f; }
  const float scale = 0.125f;                     // 1/sqrt(64)
  const float* maskb = maskf + (size_t)b * TS;

  int kkey = tid >> 2;                            // 32 keys x 4 threads
  int kdkc = (tid & 3) * 16;

  for (int j = 0; j < TS; j += 32) {
    __syncthreads();
    {
      const _Float16* base = QKV + ((size_t)b * TS + j + kkey) * QKVN + h * TDK + kdkc;
      const _Float16* gk = base + TD;             // K slice
      const _Float16* gv = base + 2 * TD;         // V slice
      *(v8h*)(sK + kkey * 64 + kdkc)     = *(const v8h*)gk;
      *(v8h*)(sK + kkey * 64 + kdkc + 8) = *(const v8h*)(gk + 8);
      v8h va = *(const v8h*)gv;
      v8h vb = *(const v8h*)(gv + 8);
#pragma unroll
      for (int i = 0; i < 8; ++i) {
        sV[(kdkc + i) * 32 + kkey]     = va[i];
        sV[(kdkc + 8 + i) * 32 + kkey] = vb[i];
      }
    }
    __syncthreads();

    // scores: 4 WMMA (2 key tiles x 2 dk chunks)
    v8f s0 = {}, s1 = {};
    {
      HV bfrag;
      bfrag.h[0] = *(const v8h*)(sK + laneN * 64 + kb);
      bfrag.h[1] = *(const v8h*)(sK + laneN * 64 + 16 + kb);
      s0 = __builtin_amdgcn_wmma_f32_16x16x32_f16(false, q0.v, false, bfrag.v, (short)0, s0, false, false);
      bfrag.h[0] = *(const v8h*)(sK + laneN * 64 + 32 + kb);
      bfrag.h[1] = *(const v8h*)(sK + laneN * 64 + 32 + 16 + kb);
      s0 = __builtin_amdgcn_wmma_f32_16x16x32_f16(false, q1.v, false, bfrag.v, (short)0, s0, false, false);
      bfrag.h[0] = *(const v8h*)(sK + (16 + laneN) * 64 + kb);
      bfrag.h[1] = *(const v8h*)(sK + (16 + laneN) * 64 + 16 + kb);
      s1 = __builtin_amdgcn_wmma_f32_16x16x32_f16(false, q0.v, false, bfrag.v, (short)0, s1, false, false);
      bfrag.h[0] = *(const v8h*)(sK + (16 + laneN) * 64 + 32 + kb);
      bfrag.h[1] = *(const v8h*)(sK + (16 + laneN) * 64 + 32 + 16 + kb);
      s1 = __builtin_amdgcn_wmma_f32_16x16x32_f16(false, q1.v, false, bfrag.v, (short)0, s1, false, false);
    }

    // online softmax (rows span 16-lane halves of the wave32 C layout)
    float mk0 = maskb[j + laneN];
    float mk1 = maskb[j + 16 + laneN];
    float sl[8], sh[8], alpha[8];
#pragma unroll
    for (int r = 0; r < 8; ++r) {
      sl[r] = s0[r] * scale + mk0;
      sh[r] = s1[r] * scale + mk1;
      float mb = red_max16(fmaxf(sl[r], sh[r]));
      float mn = fmaxf(mrow[r], mb);
      alpha[r] = __expf(mrow[r] - mn);
      sl[r] = __expf(sl[r] - mn);
      sh[r] = __expf(sh[r] - mn);
      float lb = red_sum16(sl[r] + sh[r]);
      lrow[r] = lrow[r] * alpha[r] + lb;
      mrow[r] = mn;
    }

    _Float16* P = sP[wave];
#pragma unroll
    for (int r = 0; r < 8; ++r) {
      int row = laneHi * 8 + r;
      P[row * 32 + laneN]      = (_Float16)sl[r];
      P[row * 32 + 16 + laneN] = (_Float16)sh[r];
    }
#pragma unroll
    for (int r = 0; r < 8; ++r) {
      o0[r] *= alpha[r]; o1[r] *= alpha[r]; o2[r] *= alpha[r]; o3[r] *= alpha[r];
    }
    HV pa;
    pa.h[0] = *(const v8h*)(P + laneN * 32 + kb);
    pa.h[1] = *(const v8h*)(P + laneN * 32 + 16 + kb);
    {
      HV vf;
      vf.h[0] = *(const v8h*)(sV + (0 * 16 + laneN) * 32 + kb);
      vf.h[1] = *(const v8h*)(sV + (0 * 16 + laneN) * 32 + 16 + kb);
      o0 = __builtin_amdgcn_wmma_f32_16x16x32_f16(false, pa.v, false, vf.v, (short)0, o0, false, false);
      vf.h[0] = *(const v8h*)(sV + (1 * 16 + laneN) * 32 + kb);
      vf.h[1] = *(const v8h*)(sV + (1 * 16 + laneN) * 32 + 16 + kb);
      o1 = __builtin_amdgcn_wmma_f32_16x16x32_f16(false, pa.v, false, vf.v, (short)0, o1, false, false);
      vf.h[0] = *(const v8h*)(sV + (2 * 16 + laneN) * 32 + kb);
      vf.h[1] = *(const v8h*)(sV + (2 * 16 + laneN) * 32 + 16 + kb);
      o2 = __builtin_amdgcn_wmma_f32_16x16x32_f16(false, pa.v, false, vf.v, (short)0, o2, false, false);
      vf.h[0] = *(const v8h*)(sV + (3 * 16 + laneN) * 32 + kb);
      vf.h[1] = *(const v8h*)(sV + (3 * 16 + laneN) * 32 + 16 + kb);
      o3 = __builtin_amdgcn_wmma_f32_16x16x32_f16(false, pa.v, false, vf.v, (short)0, o3, false, false);
    }
  }

#pragma unroll
  for (int r = 0; r < 8; ++r) {
    float inv = 1.0f / lrow[r];
    int s = qBase + laneHi * 8 + r;
    _Float16* op = O + ((size_t)b * TS + s) * TD + h * TDK + laneN;
    op[0]  = (_Float16)(o0[r] * inv);
    op[16] = (_Float16)(o1[r] * inv);
    op[32] = (_Float16)(o2[r] * inv);
    op[48] = (_Float16)(o3[r] * inv);
  }
}

// ---------------------------------------------------------------------------
// Fused residual add + LayerNorm. One 768-wide row per block (256 thr).
// ---------------------------------------------------------------------------
__global__ __launch_bounds__(256)
void ln_kernel(const float* __restrict__ a, const float* __restrict__ bres,
               const float* __restrict__ gamma, const float* __restrict__ beta,
               float* __restrict__ outf, _Float16* __restrict__ outh) {
  __shared__ float r1[256], r2[256];
  int row = blockIdx.x, tid = threadIdx.x;
  const float* ap = a + (size_t)row * TD;
  const float* bp = bres + (size_t)row * TD;
  float loc[3], s = 0.0f, s2 = 0.0f;
#pragma unroll
  for (int i = 0; i < 3; ++i) {
    int d = tid + i * 256;
    float v = ap[d] + bp[d];
    loc[i] = v; s += v; s2 += v * v;
  }
  r1[tid] = s; r2[tid] = s2;
  __syncthreads();
  for (int o = 128; o > 0; o >>= 1) {
    if (tid < o) { r1[tid] += r1[tid + o]; r2[tid] += r2[tid + o]; }
    __syncthreads();
  }
  float mu = r1[0] * (1.0f / TD);
  float var = r2[0] * (1.0f / TD) - mu * mu;
  float rs = rsqrtf(var + 1e-5f);
#pragma unroll
  for (int i = 0; i < 3; ++i) {
    int d = tid + i * 256;
    float v = (loc[i] - mu) * rs * gamma[d] + beta[d];
    outf[(size_t)row * TD + d] = v;
    outh[(size_t)row * TD + d] = (_Float16)v;
  }
}

// ---------------------------------------------------------------------------
// Final mean over sequence: out[b,d] = mean_s x[b,s,d]
// ---------------------------------------------------------------------------
__global__ __launch_bounds__(256)
void mean_kernel(const float* __restrict__ x, float* __restrict__ out) {
  int idx = blockIdx.x * 256 + threadIdx.x;
  if (idx >= TB * TD) return;
  int b = idx / TD, d = idx % TD;
  float s = 0.0f;
  for (int t = 0; t < TS; ++t) s += x[((size_t)b * TS + t) * TD + d];
  out[idx] = s * (1.0f / TS);
}

// ---------------------------------------------------------------------------
// Host orchestration
// ---------------------------------------------------------------------------
extern "C" void kernel_launch(void* const* d_in, const int* in_sizes, int n_in,
                              void* d_out, int out_size, void* d_ws, size_t ws_size,
                              hipStream_t stream) {
  const int*   ids   = (const int*)  d_in[0];
  const float* emb   = (const float*)d_in[1];
  const float* Wq    = (const float*)d_in[2];
  const float* bq    = (const float*)d_in[3];
  const float* Wk    = (const float*)d_in[4];
  const float* bk    = (const float*)d_in[5];
  const float* Wv    = (const float*)d_in[6];
  const float* bv    = (const float*)d_in[7];
  const float* Wo    = (const float*)d_in[8];
  const float* bo    = (const float*)d_in[9];
  const float* W1    = (const float*)d_in[10];
  const float* b1    = (const float*)d_in[11];
  const float* W2    = (const float*)d_in[12];
  const float* b2    = (const float*)d_in[13];
  const float* gamma = (const float*)d_in[14];
  const float* beta  = (const float*)d_in[15];
  float* out = (float*)d_out;

  char* wsp = (char*)d_ws;
  auto alloc = [&](size_t bytes) -> void* {
    void* r = (void*)wsp;
    wsp += (bytes + 255) & ~(size_t)255;
    return r;
  };
  const size_t NDD  = (size_t)TL * TD * TD;
  const size_t NDF  = (size_t)TL * TD * TF;
  _Float16* wqkvT = (_Float16*)alloc((size_t)TL * QKVN * TD * 2); // [L][2304][768]
  _Float16* woT   = (_Float16*)alloc(NDD * 2);                    // [L][768][768]
  _Float16* w1T   = (_Float16*)alloc(NDF * 2);                    // [L][3072][768]
  _Float16* w2T   = (_Float16*)alloc(NDF * 2);                    // [L][768][3072]
  float*    bqkv  = (float*)   alloc((size_t)TL * QKVN * 4);
  float*    xf32  = (float*)   alloc((size_t)TM * TD * 4);
  float*    x1f32 = (float*)   alloc((size_t)TM * TD * 4);
  float*    gf32  = (float*)   alloc((size_t)TM * TD * 4);
  _Float16* xf16  = (_Float16*)alloc((size_t)TM * TD * 2);
  _Float16* x1f16 = (_Float16*)alloc((size_t)TM * TD * 2);
  _Float16* qkv16 = (_Float16*)alloc((size_t)TM * QKVN * 2);
  _Float16* af16  = (_Float16*)alloc((size_t)TM * TD * 2);
  _Float16* hf16  = (_Float16*)alloc((size_t)TM * TF * 2);
  float*    maskf = (float*)   alloc((size_t)TM * 4);

  // weight transposition / fusion (every call: deterministic)
  {
    int nDD = (int)NDD, nDF = (int)NDF;
    int gDD = (nDD + 255) / 256, gDF = (nDF + 255) / 256;
    cvt_transpose_kernel<<<gDD, 256, 0, stream>>>(Wq, wqkvT, TD, TD, QKVN * TD, 0,      nDD);
    cvt_transpose_kernel<<<gDD, 256, 0, stream>>>(Wk, wqkvT, TD, TD, QKVN * TD, TD,     nDD);
    cvt_transpose_kernel<<<gDD, 256, 0, stream>>>(Wv, wqkvT, TD, TD, QKVN * TD, 2 * TD, nDD);
    cvt_transpose_kernel<<<gDD, 256, 0, stream>>>(Wo, woT, TD, TD, TD * TD, 0, nDD);
    cvt_transpose_kernel<<<gDF, 256, 0, stream>>>(W1, w1T, TD, TF, TF * TD, 0, nDF);
    cvt_transpose_kernel<<<gDF, 256, 0, stream>>>(W2, w2T, TF, TD, TD * TF, 0, nDF);
    catbias_kernel<<<(TL * QKVN + 255) / 256, 256, 0, stream>>>(bq, bk, bv, bqkv);
  }
  // embedding + PE + mask
  {
    int n = TM * TD;
    embed_kernel<<<(n + 255) / 256, 256, 0, stream>>>(ids, emb, xf32, xf16, maskf);
  }

  dim3 gQKV (QKVN / 128, TM / 128);
  dim3 gProj(TD   / 128, TM / 128);
  dim3 gFF1 (TF   / 128, TM / 128);
  dim3 gAttn(TS / 64, TH, TB);

  for (int l = 0; l < TL; ++l) {
    const _Float16* wqkvl = wqkvT + (size_t)l * QKVN * TD;
    const _Float16* wol   = woT   + (size_t)l * TD * TD;
    const _Float16* w1l   = w1T   + (size_t)l * TF * TD;
    const _Float16* w2l   = w2T   + (size_t)l * TD * TF;
    const float* bqkvl = bqkv + (size_t)l * QKVN;
    const float* bol = bo + (size_t)l * TD;
    const float* b1l = b1 + (size_t)l * TF;
    const float* b2l = b2 + (size_t)l * TD;
    const float* gl  = gamma + (size_t)l * TD;
    const float* btl = beta  + (size_t)l * TD;

    // fused QKV projection (f16 out, [M][2304])
    gemm_f16_kernel<<<gQKV, 256, 0, stream>>>(xf16, wqkvl, bqkvl, nullptr, qkv16, TM, QKVN, TD, 0);
    // flash attention
    attn_kernel<<<gAttn, 128, 0, stream>>>(qkv16, maskf, af16);
    // output projection (f32 out)
    gemm_f16_kernel<<<gProj, 256, 0, stream>>>(af16, wol, bol, gf32, nullptr, TM, TD, TD, 0);
    // x1 = LN(x + attn_out)
    ln_kernel<<<TM, 256, 0, stream>>>(xf32, gf32, gl, btl, x1f32, x1f16);
    // FFN up + ReLU (f16 out)
    gemm_f16_kernel<<<gFF1, 256, 0, stream>>>(x1f16, w1l, b1l, nullptr, hf16, TM, TF, TD, 1);
    // FFN down (f32 out)
    gemm_f16_kernel<<<gProj, 256, 0, stream>>>(hf16, w2l, b2l, gf32, nullptr, TM, TD, TF, 0);
    // x = LN(x1 + ff)
    ln_kernel<<<TM, 256, 0, stream>>>(x1f32, gf32, gl, btl, xf32, xf16);
  }

  mean_kernel<<<(TB * TD + 255) / 256, 256, 0, stream>>>(xf32, out);
}
